// BertGCN_4544075399734
// MI455X (gfx1250) — matrix operand order repaired and checked by reading, not compiled, not run on hardware
//
#include <hip/hip_runtime.h>

// ---------------------------------------------------------------------------
// BertGCN forward on MI455X (gfx1250): bf16 WMMA GEMMs, f32 accumulate,
// double-buffered LDS pipeline, async global->LDS for the bf16 B operand,
// pointer-increment addressing (no per-iteration 64-bit multiplies).
//   GEMM1: XW^T   = (label_features @ gc_weight)^T            (bf16, ws)
//   GEMM2: lembed = relu(edges @ XW + gc_bias)                (f32,  ws)
//   diag : diag[l]= sum_h lembed[l,h] * clf_weight[l,h]       (f32,  ws)
//   GEMM3: out    = bert_cls @ clf_weight^T + diag + clf_bias (f32, d_out)
// ---------------------------------------------------------------------------

typedef __bf16 v16bf __attribute__((ext_vector_type(16)));
typedef __bf16 v8bf  __attribute__((ext_vector_type(8)));
typedef __bf16 v4bf  __attribute__((ext_vector_type(4)));
typedef float  v8f   __attribute__((ext_vector_type(8)));
typedef float  v4f   __attribute__((ext_vector_type(4)));
typedef int    v2i   __attribute__((ext_vector_type(2)));

// pointer-to-<2 x i32> in global (AS1) / LDS (AS3) address spaces, the
// payload type __builtin_amdgcn_global_load_async_to_lds_b64 expects
typedef __attribute__((address_space(1))) v2i* as1_v2i;
typedef __attribute__((address_space(3))) v2i* as3_v2i;

#if __has_builtin(__builtin_amdgcn_global_load_async_to_lds_b64)
#  define HAVE_ASYNC_LDS 1
#else
#  define HAVE_ASYNC_LDS 0
#endif

__device__ __forceinline__ void wait_asynccnt0() {
#if HAVE_ASYNC_LDS
#  if __has_builtin(__builtin_amdgcn_s_wait_asynccnt)
    __builtin_amdgcn_s_wait_asynccnt(0);
#  else
    asm volatile("s_wait_asynccnt 0x0" ::: "memory");
#  endif
#endif
}

union FragU {                 // two 16B LDS loads -> one WMMA operand
    uint4 q[2];
    v16bf v;
};

#define BM 128
#define BN 128
#define BK 32
#define LDSS 40               // 32 bf16 + 8 pad => 80B rows, 16B-aligned chunks

// BMODE: 0 = B is f32 [K,N] (transpose while staging)
//        1 = B is bf16 [N,K] (k-contiguous; async copy straight into LDS)
//        2 = B is f32  [N,K] (k-contiguous, convert while staging)
template<int BMODE, bool RELU, bool OUT_BF16_T>
__global__ __launch_bounds__(256)
void gemm_bf16_wmma(const float* __restrict__ A,
                    const float* __restrict__ Bf,
                    const __bf16* __restrict__ Bh,
                    float* __restrict__ C,
                    __bf16* __restrict__ CbfT,
                    const float* __restrict__ bias1,
                    const float* __restrict__ bias2,
                    int M, int N, int K, int lda, int ldb, int ldc)
{
    __shared__ __align__(16) __bf16 sA[2][BM * LDSS];
    __shared__ __align__(16) __bf16 sB[2][BN * LDSS];

    const int tid  = threadIdx.x;
    const int lane = tid & 31;
    const int wid  = tid >> 5;      // 8 waves
    const int half = lane >> 4;     // hi/lo 16 lanes
    const int l16  = lane & 15;
    const int wm   = wid & 3;       // 4 waves x 32 rows
    const int wn   = wid >> 2;      // 2 waves x 64 cols
    const int m0 = blockIdx.y * BM;
    const int n0 = blockIdx.x * BN;

    const int rowA = tid >> 3;      // 0..31 (x4 r-steps => 128 rows)
    const int k4   = (tid & 7) * 4; // 0..28

    v8f acc[2][4];
    const v8f vzero = {0.f, 0.f, 0.f, 0.f, 0.f, 0.f, 0.f, 0.f};
#pragma unroll
    for (int mi = 0; mi < 2; ++mi)
#pragma unroll
        for (int ni = 0; ni < 4; ++ni) acc[mi][ni] = vzero;

    v4f aR[4];                      // A prefetch regs (all modes)
    v4f bR[4];                      // B prefetch regs (modes 0/2, 1-fallback)

    // ---- loop-invariant per-thread pointers, bumped by constant strides ---
    const float* aP[4];
#pragma unroll
    for (int r = 0; r < 4; ++r)
        aP[r] = A + (size_t)(m0 + r * 32 + rowA) * lda + k4;

    const float*  bPf[4] = {nullptr, nullptr, nullptr, nullptr};
    const __bf16* bPh[4] = {nullptr, nullptr, nullptr, nullptr};
    if (BMODE == 0) {
#pragma unroll
        for (int r = 0; r < 2; ++r)
            bPf[r] = Bf + (size_t)(r * 16 + (tid >> 4)) * ldb + n0 + (tid & 15) * 8;
    } else if (BMODE == 1) {
#pragma unroll
        for (int r = 0; r < 4; ++r)
            bPh[r] = Bh + (size_t)(n0 + r * 32 + rowA) * ldb + k4;
    } else {
#pragma unroll
        for (int r = 0; r < 4; ++r)
            bPf[r] = Bf + (size_t)(n0 + r * 32 + rowA) * ldb + k4;
    }

    // ---------------- staging helpers -------------------------------------
    auto loadA = [&]() {
#pragma unroll
        for (int r = 0; r < 4; ++r) {
            aR[r] = *reinterpret_cast<const v4f*>(aP[r]);
            aP[r] += BK;
        }
    };
    auto storeA = [&](int buf) {
#pragma unroll
        for (int r = 0; r < 4; ++r)
            *reinterpret_cast<v4bf*>(&sA[buf][(r * 32 + rowA) * LDSS + k4]) =
                __builtin_convertvector(aR[r], v4bf);
    };
    auto loadB = [&](int buf) {
        if (BMODE == 0) {           // f32 [K,N]
#pragma unroll
            for (int r = 0; r < 2; ++r) {
                bR[2 * r + 0] = *reinterpret_cast<const v4f*>(bPf[r]);
                bR[2 * r + 1] = *reinterpret_cast<const v4f*>(bPf[r] + 4);
                bPf[r] += (size_t)BK * ldb;
            }
        } else if (BMODE == 1) {    // bf16 [N,K] -> async straight into LDS
#if HAVE_ASYNC_LDS
#pragma unroll
            for (int r = 0; r < 4; ++r) {
                const int n = r * 32 + rowA;
                __builtin_amdgcn_global_load_async_to_lds_b64(
                    (as1_v2i)(bPh[r]),
                    (as3_v2i)(&sB[buf][n * LDSS + k4]),
                    0, 0);
                bPh[r] += BK;
            }
#else
#pragma unroll
            for (int r = 0; r < 4; ++r) {
                bR[r] = __builtin_bit_cast(v4f,
                    *reinterpret_cast<const ulong2*>(bPh[r]));
                bPh[r] += BK;
            }
#endif
        } else {                    // f32 [N,K]
#pragma unroll
            for (int r = 0; r < 4; ++r) {
                bR[r] = *reinterpret_cast<const v4f*>(bPf[r]);
                bPf[r] += BK;
            }
        }
        (void)buf;
    };
    auto storeB = [&](int buf) {
        if (BMODE == 0) {
#pragma unroll
            for (int r = 0; r < 2; ++r) {
                const int k  = r * 16 + (tid >> 4);
                const int nq = (tid & 15) * 8;
                sB[buf][(nq + 0) * LDSS + k] = (__bf16)bR[2 * r + 0].x;
                sB[buf][(nq + 1) * LDSS + k] = (__bf16)bR[2 * r + 0].y;
                sB[buf][(nq + 2) * LDSS + k] = (__bf16)bR[2 * r + 0].z;
                sB[buf][(nq + 3) * LDSS + k] = (__bf16)bR[2 * r + 0].w;
                sB[buf][(nq + 4) * LDSS + k] = (__bf16)bR[2 * r + 1].x;
                sB[buf][(nq + 5) * LDSS + k] = (__bf16)bR[2 * r + 1].y;
                sB[buf][(nq + 6) * LDSS + k] = (__bf16)bR[2 * r + 1].z;
                sB[buf][(nq + 7) * LDSS + k] = (__bf16)bR[2 * r + 1].w;
            }
        } else if (BMODE == 1) {
#if !HAVE_ASYNC_LDS
#pragma unroll
            for (int r = 0; r < 4; ++r)
                *reinterpret_cast<ulong2*>(&sB[buf][(r * 32 + rowA) * LDSS + k4]) =
                    __builtin_bit_cast(ulong2, bR[r]);
#endif
        } else {
#pragma unroll
            for (int r = 0; r < 4; ++r)
                *reinterpret_cast<v4bf*>(&sB[buf][(r * 32 + rowA) * LDSS + k4]) =
                    __builtin_convertvector(bR[r], v4bf);
        }
    };

    // ---------------- 2-deep software pipeline ----------------------------
    const int S = K / BK;
    loadA();
    loadB(0);
    storeA(0);
    storeB(0);
    if (BMODE == 1) wait_asynccnt0();
    __syncthreads();

    for (int s = 0; s < S; ++s) {
        const int cur = s & 1;
        const int nxt = cur ^ 1;
        const bool has_next = (s + 1) < S;

        if (has_next) { loadA(); loadB(nxt); }

        // fragments (ISA 16-bit A 16x32 / B 32x16 VGPR layouts)
        FragU af[2], bf[4];
#pragma unroll
        for (int mi = 0; mi < 2; ++mi) {
            const int row = (wm * 32 + mi * 16 + l16) * LDSS;
            af[mi].q[0] = *reinterpret_cast<const uint4*>(&sA[cur][row + half * 8]);
            af[mi].q[1] = *reinterpret_cast<const uint4*>(&sA[cur][row + 16 + half * 8]);
        }
#pragma unroll
        for (int ni = 0; ni < 4; ++ni) {
            const int row = (wn * 64 + ni * 16 + l16) * LDSS;
            bf[ni].q[0] = *reinterpret_cast<const uint4*>(&sB[cur][row + half * 8]);
            bf[ni].q[1] = *reinterpret_cast<const uint4*>(&sB[cur][row + 16 + half * 8]);
        }

        if (has_next) { storeA(nxt); storeB(nxt); }

#pragma unroll
        for (int mi = 0; mi < 2; ++mi)
#pragma unroll
            for (int ni = 0; ni < 4; ++ni)
                acc[mi][ni] = __builtin_amdgcn_wmma_f32_16x16x32_bf16(
                    false, af[mi].v, false, bf[ni].v,
                    (short)0, acc[mi][ni], false, false);

        if (BMODE == 1 && has_next) wait_asynccnt0();
        __syncthreads();
    }

    // ---------------- epilogue --------------------------------------------
#pragma unroll
    for (int mi = 0; mi < 2; ++mi) {
#pragma unroll
        for (int ni = 0; ni < 4; ++ni) {
            const int row0 = m0 + wm * 32 + mi * 16 + half * 8;
            const int col  = n0 + wn * 64 + ni * 16 + l16;
            if (OUT_BF16_T) {
                *reinterpret_cast<v8bf*>(CbfT + (size_t)col * M + row0) =
                    __builtin_convertvector(acc[mi][ni], v8bf);
            } else {
                float b = 0.f;
                if (bias1) b += bias1[col];
                if (bias2) b += bias2[col];
                float* cp = C + (size_t)row0 * ldc + col;
#pragma unroll
                for (int j = 0; j < 8; ++j) {
                    float v = acc[mi][ni][j] + b;
                    if (RELU) v = v > 0.f ? v : 0.f;
                    *cp = v;
                    cp += ldc;
                }
            }
        }
    }
}

__global__ __launch_bounds__(256)
void diag_kernel(const float* __restrict__ E, const float* __restrict__ W,
                 float* __restrict__ diag, int H)
{
    __shared__ float red[256];
    const int l = blockIdx.x;
    float s = 0.f;
    for (int h = threadIdx.x; h < H; h += 256)
        s += E[(size_t)l * H + h] * W[(size_t)l * H + h];
    red[threadIdx.x] = s;
    __syncthreads();
    for (int off = 128; off > 0; off >>= 1) {
        if (threadIdx.x < off) red[threadIdx.x] += red[threadIdx.x + off];
        __syncthreads();
    }
    if (threadIdx.x == 0) diag[l] = red[0];
}

extern "C" void kernel_launch(void* const* d_in, const int* in_sizes, int n_in,
                              void* d_out, int out_size, void* d_ws, size_t ws_size,
                              hipStream_t stream)
{
    constexpr int Bq = 2048, H = 1024, L = 8192, F = 1024;

    const float* bert_cls       = (const float*)d_in[0];
    const float* label_features = (const float*)d_in[1];
    const float* edges          = (const float*)d_in[2];
    const float* gc_weight      = (const float*)d_in[3];
    const float* gc_bias        = (const float*)d_in[4];
    const float* clf_weight     = (const float*)d_in[5];
    const float* clf_bias       = (const float*)d_in[6];

    // workspace layout
    __bf16* XWt        = (__bf16*)d_ws;                                  // [H][L] bf16, 16 MB
    float* label_embed = (float*)((char*)d_ws + (size_t)H * L * 2);      // [L][H] f32, 32 MB
    float* diag        = (float*)((char*)d_ws + (size_t)H * L * 2
                                              + (size_t)L * H * 4);      // [L] f32

    // GEMM1: XW^T (bf16) = (label_features[L,F] @ gc_weight[F,H])^T
    dim3 g1(H / BN, L / BM);
    gemm_bf16_wmma<0, false, true><<<g1, 256, 0, stream>>>(
        label_features, gc_weight, nullptr, nullptr, XWt,
        nullptr, nullptr, L, H, F, F, H, 0);

    // GEMM2: label_embed = relu(edges[L,L] @ XW[L,H] + gc_bias)
    dim3 g2(H / BN, L / BM);
    gemm_bf16_wmma<1, true, false><<<g2, 256, 0, stream>>>(
        edges, nullptr, XWt, label_embed, nullptr,
        gc_bias, nullptr, L, H, L, L, L, H);

    // diag[l] = sum_h label_embed[l,h] * clf_weight[l,h]
    diag_kernel<<<L, 256, 0, stream>>>(label_embed, clf_weight, diag, H);

    // GEMM3: out = bert_cls[B,H] @ clf_weight[L,H]^T + diag + clf_bias
    dim3 g3(L / BN, Bq / BM);
    gemm_bf16_wmma<2, false, false><<<g3, 256, 0, stream>>>(
        bert_cls, clf_weight, nullptr, (float*)d_out, nullptr,
        diag, clf_bias, Bq, L, H, H, H, L);
}
